// Head_31155692765805
// MI455X (gfx1250) — matrix operand order repaired and checked by reading, not compiled, not run on hardware
//
#include <hip/hip_runtime.h>

// ---------------------------------------------------------------------------
// Causal single-head attention, CDNA5 (gfx1250), wave32 + WMMA bf16.
//   Stage 0: wcvt     : Wk/Wq/Wv fp32 -> bf16 (once)
//   Stage 1: qkv_proj : x[B,T,C] fp32 -> K,Q [B,T,H] bf16, Vt [B,H,T] bf16
//   Stage 2: attn     : flash-attention per (batch, 16-row query tile)
// ---------------------------------------------------------------------------

typedef __attribute__((ext_vector_type(16))) __bf16 bf16x16;
typedef __attribute__((ext_vector_type(8)))  float  f32x8;
typedef __attribute__((ext_vector_type(4)))  float  f32x4;
typedef unsigned short u16;
typedef unsigned int   u32;

struct __align__(16) U4 { u32 x, y, z, w; };

union Frag {
    bf16x16 v;
    u16     u[16];
    u32     w[8];
    U4      q[2];
};

// scalar fp32 -> bf16 (round-to-nearest-even)
__device__ __forceinline__ u16 f2bf(float f) {
    u32 u = __builtin_bit_cast(u32, f);
    u += 0x7FFFu + ((u >> 16) & 1u);
    return (u16)(u >> 16);
}

// two fp32 -> packed {hi_bf16, lo_bf16} u32 via v_perm_b32
__device__ __forceinline__ u32 pk2(float lo, float hi) {
    u32 a = __builtin_bit_cast(u32, lo); a += 0x7FFFu + ((a >> 16) & 1u);
    u32 b = __builtin_bit_cast(u32, hi); b += 0x7FFFu + ((b >> 16) & 1u);
    return __builtin_amdgcn_perm(b, a, 0x07060302u);   // {b[3],b[2],a[3],a[2]}
}

#define WMMA_BF16(A, Bm, Cc) \
    __builtin_amdgcn_wmma_f32_16x16x32_bf16(false, (A), false, (Bm), (short)0, (Cc), false, false)

constexpr int NB = 512;   // batch
constexpr int NT = 256;   // sequence
constexpr int NC = 384;   // embedding
constexpr int NH = 64;    // head size

// ---------------------------------------------------------------------------
// Stage 0: convert the three weight matrices to bf16, packed [3][H][C].
// ---------------------------------------------------------------------------
__global__ __launch_bounds__(256) void wcvt(
    const float* __restrict__ Wk, const float* __restrict__ Wq,
    const float* __restrict__ Wv, u16* __restrict__ Wb)
{
    const int i = (blockIdx.x * 256 + threadIdx.x) * 4;   // 24 blocks cover 24576
    const f32x4 a = *(const f32x4*)(Wk + i);
    const f32x4 b = *(const f32x4*)(Wq + i);
    const f32x4 c = *(const f32x4*)(Wv + i);
    u32* d0 = (u32*)(Wb + i);
    u32* d1 = (u32*)(Wb + 24576 + i);
    u32* d2 = (u32*)(Wb + 49152 + i);
    d0[0] = pk2(a.x, a.y); d0[1] = pk2(a.z, a.w);
    d1[0] = pk2(b.x, b.y); d1[1] = pk2(b.z, b.w);
    d2[0] = pk2(c.x, c.y); d2[1] = pk2(c.z, c.w);
}

// ---------------------------------------------------------------------------
// Stage 1: Q/K/V projection. One wave = one 16-row tile of x.
// All 12 B-fragments load as one clause; an empty asm "pin" with tied +v
// operands forces 12 distinct register buffers and a single loadcnt drain
// per chunk (instead of 12 load->drain->wmma round trips).
// ---------------------------------------------------------------------------
__global__ __launch_bounds__(128)
__attribute__((amdgpu_waves_per_eu(2)))
void qkv_proj(
    const float* __restrict__ x,
    const u16*  __restrict__ Wb,   // [3][64][384] bf16; frag j = rows 16j..16j+15
    u16* __restrict__ Kb,          // [B,T,H] bf16
    u16* __restrict__ Qb,          // [B,T,H] bf16
    u16* __restrict__ Vt)          // [B,H,T] bf16
{
    const int wave = blockIdx.x * 4 + (threadIdx.x >> 5);
    const int lane = threadIdx.x & 31;
    const int half = lane >> 4;
    const int l16  = lane & 15;

    const int b  = wave >> 4;      // 16 row-tiles per batch (T/16)
    const int tt = wave & 15;

    const float* xrow = x + (size_t)(b * NT + tt * 16 + l16) * NC + half * 16;

    f32x8 acc[12] = {};            // acc[w*4+nt]; weight rows = 16*(w*4+nt)..

    // prologue: x floats for chunk 0
    f32x4 f0 = *(const f32x4*)(xrow);
    f32x4 f1 = *(const f32x4*)(xrow + 4);
    f32x4 f2 = *(const f32x4*)(xrow + 8);
    f32x4 f3 = *(const f32x4*)(xrow + 12);

    #pragma unroll 1
    for (int ch = 0; ch < 12; ++ch) {              // K dim: 12 x 32 = 384
        // ---- A fragment from already-loaded floats
        Frag a;
        a.w[0] = pk2(f0.x, f0.y); a.w[1] = pk2(f0.z, f0.w);
        a.w[2] = pk2(f1.x, f1.y); a.w[3] = pk2(f1.z, f1.w);
        a.w[4] = pk2(f2.x, f2.y); a.w[5] = pk2(f2.z, f2.w);
        a.w[6] = pk2(f3.x, f3.y); a.w[7] = pk2(f3.z, f3.w);

        // ---- load ALL 12 bf16 B fragments (one clause, 24 b128 loads)
        const u16* wp0 = Wb + (size_t)l16 * NC + ch * 32 + half * 16;
        Frag bm[12];
        #pragma unroll
        for (int j = 0; j < 12; ++j) {
            const u16* p = wp0 + (size_t)(16 * j) * NC;
            bm[j].q[0] = *(const U4*)p;
            bm[j].q[1] = *(const U4*)(p + 8);
        }

        // ---- software-pipeline: issue next chunk's x loads now
        if (ch + 1 < 12) {
            const float* nx = xrow + (ch + 1) * 32;
            f0 = *(const f32x4*)(nx);
            f1 = *(const f32x4*)(nx + 4);
            f2 = *(const f32x4*)(nx + 8);
            f3 = *(const f32x4*)(nx + 12);
        }

        // ---- scheduling pin: all 12 fragments materialize HERE, in
        // distinct registers; loads cannot sink below, WMMAs cannot hoist.
        asm volatile("" :
            "+v"(bm[0].v), "+v"(bm[1].v), "+v"(bm[2].v),  "+v"(bm[3].v),
            "+v"(bm[4].v), "+v"(bm[5].v), "+v"(bm[6].v),  "+v"(bm[7].v),
            "+v"(bm[8].v), "+v"(bm[9].v), "+v"(bm[10].v), "+v"(bm[11].v));

        #pragma unroll
        for (int j = 0; j < 12; ++j)
            acc[j] = WMMA_BF16(a.v, bm[j].v, acc[j]);
    }

    // ---- store: C-frag element r -> row t = tt*16 + r + 8*half, col h
    #pragma unroll
    for (int w = 0; w < 3; ++w) {
        #pragma unroll
        for (int nt = 0; nt < 4; ++nt) {
            const int h = nt * 16 + l16;
            #pragma unroll
            for (int r = 0; r < 8; ++r) {
                const int t = tt * 16 + r + half * 8;
                const u16 val = f2bf(acc[w * 4 + nt][r]);
                if (w == 2)       Vt[(size_t)(b * NH + h) * NT + t] = val;
                else if (w == 0)  Kb[(size_t)(b * NT + t) * NH + h] = val;
                else              Qb[(size_t)(b * NT + t) * NH + h] = val;
            }
        }
    }
}

// ---------------------------------------------------------------------------
// Stage 2: flash attention. One wave = one (batch, 16-query-row tile).
// Key loop steps by 32. All 8 K/V fragments load at loop top and are pinned
// into distinct registers so V-load latency hides under the softmax VALU.
// ---------------------------------------------------------------------------
__global__ __launch_bounds__(128)
__attribute__((amdgpu_waves_per_eu(2)))
void attn(
    const u16* __restrict__ Qb,
    const u16* __restrict__ Kb,
    const u16* __restrict__ Vt,
    float* __restrict__ out)
{
    __shared__ u16 pbuf[4][16 * 32];   // 1 KB P-scratch per wave

    const int wv   = threadIdx.x >> 5;
    const int wave = blockIdx.x * 4 + wv;
    const int lane = threadIdx.x & 31;
    const int half = lane >> 4;
    const int l16  = lane & 15;

    const int b  = wave >> 4;          // 16 q-tiles per batch
    const int qt = wave & 15;

    // ---- Q A-fragments for both 32-wide K-chunks of H=64 (loaded once)
    const u16* qrowp = Qb + (size_t)(b * NT + qt * 16 + l16) * NH;
    Frag a0, a1;
    a0.q[0] = *(const U4*)(qrowp + half * 16);
    a0.q[1] = *(const U4*)(qrowp + half * 16 + 8);
    a1.q[0] = *(const U4*)(qrowp + 32 + half * 16);
    a1.q[1] = *(const U4*)(qrowp + 32 + half * 16 + 8);

    float m_i[8], l_i[8];
    #pragma unroll
    for (int r = 0; r < 8; ++r) { m_i[r] = -__builtin_inff(); l_i[r] = 0.0f; }
    f32x8 o[4] = {};

    const float scale = 0.0510310363f;            // C^-0.5 (ref scales by C, not H)
    const int nIter = (qt >> 1) + 1;              // 32 keys per iter, causal bound
    u16* myp = pbuf[wv];

    #pragma unroll 1
    for (int it = 0; it < nIter; ++it) {
        const int st = it * 32;
        const u16* k0p = Kb + (size_t)(b * NT + st + l16) * NH;        // keys st..st+15
        const u16* k1p = Kb + (size_t)(b * NT + st + 16 + l16) * NH;   // keys st+16..st+31

        if (it + 1 < nIter)   // global_prefetch_b8 of next K block
            __builtin_prefetch(Kb + (size_t)(b * NT + st + 32 + l16) * NH, 0, 0);

        // ---- load ALL K and V fragments up-front
        Frag bk0, bk1, bk2, bk3;
        bk0.q[0] = *(const U4*)(k0p + half * 16);
        bk0.q[1] = *(const U4*)(k0p + half * 16 + 8);
        bk1.q[0] = *(const U4*)(k0p + 32 + half * 16);
        bk1.q[1] = *(const U4*)(k0p + 32 + half * 16 + 8);
        bk2.q[0] = *(const U4*)(k1p + half * 16);
        bk2.q[1] = *(const U4*)(k1p + half * 16 + 8);
        bk3.q[0] = *(const U4*)(k1p + 32 + half * 16);
        bk3.q[1] = *(const U4*)(k1p + 32 + half * 16 + 8);

        Frag bv[4];
        #pragma unroll
        for (int nt = 0; nt < 4; ++nt) {
            const u16* vp = Vt + (size_t)(b * NH + nt * 16 + l16) * NT + st + half * 16;
            bv[nt].q[0] = *(const U4*)(vp);
            bv[nt].q[1] = *(const U4*)(vp + 8);
        }

        // ---- scheduling pin: distinct registers, one load clause
        asm volatile("" :
            "+v"(bk0.v),   "+v"(bk1.v),   "+v"(bk2.v),   "+v"(bk3.v),
            "+v"(bv[0].v), "+v"(bv[1].v), "+v"(bv[2].v), "+v"(bv[3].v));

        // ---- S = Q K^T : two 16x16 fragments, K-dim H=64 as 2 chunks of 32
        f32x8 s0 = {}, s1 = {};
        s0 = WMMA_BF16(a0.v, bk0.v, s0);
        s0 = WMMA_BF16(a1.v, bk1.v, s0);
        s1 = WMMA_BF16(a0.v, bk2.v, s1);
        s1 = WMMA_BF16(a1.v, bk3.v, s1);

        // ---- causal mask + scale; online softmax (rows = r + 8*half)
        const int col0 = st + l16, col1 = st + 16 + l16;
        float p0[8], p1[8], rm[8];
        #pragma unroll
        for (int r = 0; r < 8; ++r) {
            const int qrow = qt * 16 + r + half * 8;
            const float v0 = (col0 <= qrow) ? s0[r] * scale : -__builtin_inff();
            const float v1 = (col1 <= qrow) ? s1[r] * scale : -__builtin_inff();
            p0[r] = v0; p1[r] = v1;
            rm[r] = fmaxf(v0, v1);
        }
        #pragma unroll
        for (int off = 8; off; off >>= 1)
            #pragma unroll
            for (int r = 0; r < 8; ++r)
                rm[r] = fmaxf(rm[r], __shfl_xor(rm[r], off, 32));

        float alpha[8], rs[8];
        #pragma unroll
        for (int r = 0; r < 8; ++r) {
            const float mn = fmaxf(m_i[r], rm[r]);
            alpha[r] = __expf(m_i[r] - mn);
            m_i[r]   = mn;
            p0[r] = __expf(p0[r] - mn);
            p1[r] = __expf(p1[r] - mn);
            rs[r] = p0[r] + p1[r];
        }
        #pragma unroll
        for (int off = 8; off; off >>= 1)
            #pragma unroll
            for (int r = 0; r < 8; ++r)
                rs[r] += __shfl_xor(rs[r], off, 32);
        #pragma unroll
        for (int r = 0; r < 8; ++r) l_i[r] = l_i[r] * alpha[r] + rs[r];

        #pragma unroll
        for (int nt = 0; nt < 4; ++nt)
            #pragma unroll
            for (int r = 0; r < 8; ++r)
                o[nt][r] *= alpha[r];

        // ---- relayout P: C-frag -> bf16 A-frag via per-wave LDS scratch
        #pragma unroll
        for (int r = 0; r < 8; ++r) {
            const int m = r + half * 8;
            myp[m * 32 + l16]      = f2bf(p0[r]);
            myp[m * 32 + 16 + l16] = f2bf(p1[r]);
        }
        asm volatile("s_wait_dscnt 0" ::: "memory");  // DS in-order per wave
        Frag pa;
        pa.q[0] = *(const U4*)(myp + l16 * 32 + half * 16);
        pa.q[1] = *(const U4*)(myp + l16 * 32 + half * 16 + 8);

        // ---- O += P V : V fragments already resident
        #pragma unroll
        for (int nt = 0; nt < 4; ++nt)
            o[nt] = WMMA_BF16(pa.v, bv[nt].v, o[nt]);
    }

    // ---- epilogue: divide by softmax denominator, store fp32
    #pragma unroll
    for (int r = 0; r < 8; ++r) {
        const float inv = 1.0f / l_i[r];
        const int t = qt * 16 + r + half * 8;
        float* op = out + (size_t)(b * NT + t) * NH;
        #pragma unroll
        for (int nt = 0; nt < 4; ++nt)
            op[nt * 16 + l16] = o[nt][r] * inv;
    }
}

// ---------------------------------------------------------------------------
extern "C" void kernel_launch(void* const* d_in, const int* in_sizes, int n_in,
                              void* d_out, int out_size, void* d_ws, size_t ws_size,
                              hipStream_t stream) {
    (void)in_sizes; (void)n_in; (void)out_size; (void)ws_size;
    const float* x  = (const float*)d_in[0];
    const float* Wk = (const float*)d_in[1];
    const float* Wq = (const float*)d_in[2];
    const float* Wv = (const float*)d_in[3];

    const size_t S = (size_t)NB * NT * NH;   // elements per bf16 buffer
    u16* Kb = (u16*)d_ws;
    u16* Qb = Kb + S;
    u16* Vt = Qb + S;
    u16* Wb = Vt + S;                        // 3*64*384 bf16 = 147 KB extra

    wcvt    <<<  24, 256, 0, stream>>>(Wk, Wq, Wv, Wb);
    qkv_proj<<<2048, 128, 0, stream>>>(x, Wb, Kb, Qb, Vt);
    attn    <<<2048, 128, 0, stream>>>(Qb, Kb, Vt, (float*)d_out);
}
